// Model_63539746177516
// MI455X (gfx1250) — matrix operand order repaired and checked by reading, not compiled, not run on hardware
//
#include <hip/hip_runtime.h>

// ---------------------------------------------------------------------------
// Problem constants (from the reference)
// ---------------------------------------------------------------------------
#define B_   16
#define S_   1024
#define D_   256
#define H_   4
#define FF_  1024
#define HD_  64          // D_/H_
#define M_   (B_ * S_)   // 16384 token rows
#define NB_  2
#define QKV_ (3 * D_)    // 768

// ---------------------------------------------------------------------------
// CDNA5 WMMA plumbing (wave32, 16x16x32 bf16 -> f32)
// ---------------------------------------------------------------------------
typedef __attribute__((ext_vector_type(16))) __bf16 v16bf;
typedef __attribute__((ext_vector_type(8)))  __bf16 v8bf;
typedef __attribute__((ext_vector_type(8)))  float  v8f;

__device__ inline v8f wmma_bf16(v16bf a, v16bf b, v8f c) {
  return __builtin_amdgcn_wmma_f32_16x16x32_bf16(
      /*neg_a=*/false, a, /*neg_b=*/false, b,
      /*c_mod=*/(short)0, c, /*reuse_a=*/false, /*reuse_b=*/false);
}

// A-matrix fragment (16x32, 16-bit): lane L holds row (L&15); its 16
// elements are K = k0+16*half+[0..15] (half = L>>4), matching the B
// fragment -> two contiguous 16-byte loads.
__device__ inline v16bf load_a16x32(const __bf16* base, int ld, int row,
                                    int k0, int lane) {
  int m    = row + (lane & 15);
  int half = lane >> 4;
  const __bf16* p = base + (size_t)m * ld + k0 + 16 * half;
  v8bf lo = *(const v8bf*)p;
  v8bf hi = *(const v8bf*)(p + 8);
  return __builtin_shufflevector(lo, hi, 0, 1, 2, 3, 4, 5, 6, 7,
                                 8, 9, 10, 11, 12, 13, 14, 15);
}

// B-matrix fragment (32x16, 16-bit): lane L holds column (L&15); its 16
// elements are K = k0 + 16*half + [0..15] -> one contiguous 32-byte load.
// `base` is laid out with B's columns as rows of leading dimension `ld`
// (i.e. B = base^T), which matches both W(N,K) for X@W^T and the
// pre-transposed V.
__device__ inline v16bf load_b32x16(const __bf16* base, int ld, int n0,
                                    int k0, int lane) {
  int n    = n0 + (lane & 15);
  int half = lane >> 4;
  return *(const v16bf*)(base + (size_t)n * ld + k0 + 16 * half);
}

// ---------------------------------------------------------------------------
// Small elementwise kernels
// ---------------------------------------------------------------------------
__global__ void convert_bf16_kernel(const float* __restrict__ src,
                                    __bf16* __restrict__ dst, int n) {
  int i = blockIdx.x * blockDim.x + threadIdx.x;
  if (i < n) dst[i] = (__bf16)src[i];
}

__global__ void embed_kernel(const int* __restrict__ tok,
                             const int* __restrict__ nup,
                             const float* __restrict__ emb,
                             float* __restrict__ feat,
                             __bf16* __restrict__ xb) {
  size_t i = (size_t)blockIdx.x * blockDim.x + threadIdx.x;
  if (i >= (size_t)M_ * D_) return;
  int m = (int)(i / D_);
  int d = (int)(i % D_);
  float sc = 1.0f + (nup[m] > 0 ? 1.0f : 0.0f);   // 1 + clip(n_upgrades,0,1)
  float v  = emb[(size_t)tok[m] * D_ + d] * sc;
  feat[i] = v;
  xb[i]   = (__bf16)v;
}

// vt[(b*H+h)*HD + d][t] = V[b, t, h, d]   (so P@V B-fragments are contiguous)
__global__ void transpose_v_kernel(const __bf16* __restrict__ qkv,
                                   __bf16* __restrict__ vt) {
  size_t i = (size_t)blockIdx.x * blockDim.x + threadIdx.x;
  if (i >= (size_t)B_ * H_ * HD_ * S_) return;
  int t = (int)(i % S_);
  int d = (int)((i / S_) % HD_);
  int h = (int)((i / ((size_t)S_ * HD_)) % H_);
  int b = (int)(i / ((size_t)S_ * HD_ * H_));
  vt[i] = qkv[((size_t)(b * S_ + t)) * QKV_ + 2 * D_ + h * HD_ + d];
}

// feat = LayerNorm(x + y) * g + beta ; also produce bf16 copy for next GEMM
__global__ __launch_bounds__(256) void add_ln_kernel(
    const float* __restrict__ x, const float* __restrict__ y,
    const float* __restrict__ g, const float* __restrict__ beta,
    float* __restrict__ feat, __bf16* __restrict__ xb) {
  __shared__ float s1[256];
  __shared__ float s2[256];
  int m = blockIdx.x, d = threadIdx.x;
  size_t idx = (size_t)m * D_ + d;
  float v = x[idx] + y[idx];
  s1[d] = v;
  s2[d] = v * v;
  __syncthreads();
  for (int off = 128; off > 0; off >>= 1) {
    if (d < off) { s1[d] += s1[d + off]; s2[d] += s2[d + off]; }
    __syncthreads();
  }
  float mean = s1[0] * (1.0f / D_);
  float var  = s2[0] * (1.0f / D_) - mean * mean;
  float o = (v - mean) * rsqrtf(var + 1e-5f) * g[d] + beta[d];
  feat[idx] = o;
  xb[idx]   = (__bf16)o;
}

// logits = feat @ proj_w^T + proj_b  (N = 2, pure bandwidth)
__global__ void proj_kernel(const float* __restrict__ feat,
                            const float* __restrict__ pw,
                            const float* __restrict__ pb,
                            float* __restrict__ out) {
  int m = blockIdx.x * blockDim.x + threadIdx.x;
  if (m >= M_) return;
  float a0 = pb[0], a1 = pb[1];
  const float* f = feat + (size_t)m * D_;
  for (int d = 0; d < D_; ++d) {
    float x = f[d];
    a0 += x * pw[d];
    a1 += x * pw[D_ + d];
  }
  out[2 * m + 0] = a0;
  out[2 * m + 1] = a1;
}

// ---------------------------------------------------------------------------
// Dense GEMM:  Y[M,N] = X[M,K] @ W[N,K]^T + bias  (bf16 in, f32 accum)
// Register-blocked: one wave32 owns a 16x64 output strip (4 accumulators),
// so each A fragment load feeds 4 WMMAs -> 2.5 b128 loads per WMMA and a
// 4x smaller grid-level A re-read factor.  All N dims here are % 64 == 0.
// ---------------------------------------------------------------------------
__global__ __launch_bounds__(256) void gemm_wmma_kernel(
    const __bf16* __restrict__ X, const __bf16* __restrict__ W,
    const float* __restrict__ bias, float* __restrict__ Yf,
    __bf16* __restrict__ Yb, int Mrows, int N, int K, int relu) {
  int wave  = (blockIdx.x * blockDim.x + threadIdx.x) >> 5;
  int ngrp  = N >> 6;                       // groups of 4 n-tiles (64 cols)
  int total = (Mrows >> 4) * ngrp;
  if (wave >= total) return;
  int tm = wave / ngrp;
  int n0 = (wave % ngrp) << 6;
  int lane = threadIdx.x & 31;
  int half = lane >> 4, lw = lane & 15;

  v8f acc[4];
#pragma unroll
  for (int j = 0; j < 4; ++j) {
    float bz = bias ? bias[n0 + j * 16 + lw] : 0.0f;
    acc[j] = (v8f){bz, bz, bz, bz, bz, bz, bz, bz};
  }

  for (int k0 = 0; k0 < K; k0 += 32) {
    v16bf a = load_a16x32(X, K, tm * 16, k0, lane);
#pragma unroll
    for (int j = 0; j < 4; ++j) {
      v16bf b = load_b32x16(W, K, n0 + j * 16, k0, lane);
      acc[j] = wmma_bf16(a, b, acc[j]);
    }
  }

#pragma unroll
  for (int j = 0; j < 4; ++j) {
#pragma unroll
    for (int v = 0; v < 8; ++v) {  // C/D layout: row = v + 8*half, col = lw
      float val = acc[j][v];
      if (relu) val = fmaxf(val, 0.0f);
      size_t idx = (size_t)(tm * 16 + v + 8 * half) * N + n0 + j * 16 + lw;
      if (Yf) Yf[idx] = val;
      if (Yb) Yb[idx] = (__bf16)val;
    }
  }
}

// ---------------------------------------------------------------------------
// Fused flash attention. One wave32 per (batch, head, 16-query tile).
// Streams K/V in 32-token chunks: 4 WMMA for Q.K^T, online softmax with
// __shfl_xor row reductions, P re-layout through a per-wave LDS tile
// (s_wait_dscnt for the in-wave store->load turnaround), 4 WMMA for P.V.
// ---------------------------------------------------------------------------
__device__ inline bool mask_allowed(int q, int k, int npad, int cstart) {
  bool pad_ok    = (q < npad) && (k == q);
  bool deck_ok   = (q >= npad) && (q < cstart) && (k >= npad) && (k < cstart);
  bool choice_ok = (q >= cstart) && (k >= npad);  // MASK_INTER_CHOICES == False
  return pad_ok || deck_ok || choice_ok;
}

__global__ __launch_bounds__(256) void attention_kernel(
    const __bf16* __restrict__ qkv, const __bf16* __restrict__ vt,
    const int* __restrict__ n_pad, const int* __restrict__ n_choice,
    __bf16* __restrict__ out) {
  __shared__ alignas(32) __bf16 ptile[8][16 * 32];  // per-wave P scratch

  int wave   = (blockIdx.x * blockDim.x + threadIdx.x) >> 5;
  int qtiles = S_ / 16;
  if (wave >= B_ * H_ * qtiles) return;
  int qt = wave % qtiles;
  int h  = (wave / qtiles) % H_;
  int b  = wave / (qtiles * H_);
  int lane = threadIdx.x & 31;
  int half = lane >> 4, lw = lane & 15;
  __bf16* pt = ptile[(threadIdx.x >> 5) & 7];

  int npad   = n_pad[b];
  int cstart = S_ - n_choice[b];

  const __bf16* qbase = qkv + (size_t)b * S_ * QKV_ + h * HD_;
  const __bf16* kbase = qkv + (size_t)b * S_ * QKV_ + D_ + h * HD_;
  const __bf16* vbase = vt + (size_t)(b * H_ + h) * HD_ * S_;  // [d][t]

  // Q fragments are loop-invariant: 16 queries x 64 head-dims.
  v16bf qa0 = load_a16x32(qbase, QKV_, qt * 16, 0, lane);
  v16bf qa1 = load_a16x32(qbase, QKV_, qt * 16, 32, lane);

  v8f acc0 = {}, acc1 = {}, acc2 = {}, acc3 = {};
  float mrow[8], lrow[8];
#pragma unroll
  for (int v = 0; v < 8; ++v) { mrow[v] = -1e30f; lrow[v] = 0.0f; }

  for (int kt0 = 0; kt0 < S_; kt0 += 32) {
    // ---- scores: 16 queries x 32 keys, contracted over HD=64 ----
    v8f z = {};
    v8f s0 = wmma_bf16(qa0, load_b32x16(kbase, QKV_, kt0, 0, lane), z);
    s0 = wmma_bf16(qa1, load_b32x16(kbase, QKV_, kt0, 32, lane), s0);
    v8f s1 = wmma_bf16(qa0, load_b32x16(kbase, QKV_, kt0 + 16, 0, lane), z);
    s1 = wmma_bf16(qa1, load_b32x16(kbase, QKV_, kt0 + 16, 32, lane), s1);

    // ---- mask + online softmax (row = v + 8*half, col = lw / lw+16) ----
#pragma unroll
    for (int v = 0; v < 8; ++v) {
      int q   = qt * 16 + v + 8 * half;
      int k0t = kt0 + lw;
      int k1t = kt0 + 16 + lw;
      float x0 = mask_allowed(q, k0t, npad, cstart) ? s0[v] * 0.125f : -1e30f;
      float x1 = mask_allowed(q, k1t, npad, cstart) ? s1[v] * 0.125f : -1e30f;
      float tmax = fmaxf(x0, x1);
      for (int off = 1; off < 16; off <<= 1)
        tmax = fmaxf(tmax, __shfl_xor(tmax, off, 32));
      float mn = fmaxf(mrow[v], tmax);
      float p0 = __expf(x0 - mn);
      float p1 = __expf(x1 - mn);
      float ts = p0 + p1;
      for (int off = 1; off < 16; off <<= 1) ts += __shfl_xor(ts, off, 32);
      float sc = __expf(mrow[v] - mn);
      lrow[v] = lrow[v] * sc + ts;
      mrow[v] = mn;
      acc0[v] *= sc; acc1[v] *= sc; acc2[v] *= sc; acc3[v] *= sc;
      pt[(v + 8 * half) * 32 + lw]      = (__bf16)p0;
      pt[(v + 8 * half) * 32 + 16 + lw] = (__bf16)p1;
    }
    // LDS is processed in-order per wave; wait for the stores, then re-read
    // the 16x32 P tile in A-fragment layout.
    asm volatile("s_wait_dscnt 0" ::: "memory");
    v16bf pa = load_a16x32(pt, 32, 0, 0, lane);

    // ---- out += P @ V : 4 n-tiles across HD=64 ----
    acc0 = wmma_bf16(pa, load_b32x16(vbase, S_, 0,  kt0, lane), acc0);
    acc1 = wmma_bf16(pa, load_b32x16(vbase, S_, 16, kt0, lane), acc1);
    acc2 = wmma_bf16(pa, load_b32x16(vbase, S_, 32, kt0, lane), acc2);
    acc3 = wmma_bf16(pa, load_b32x16(vbase, S_, 48, kt0, lane), acc3);
  }

#pragma unroll
  for (int v = 0; v < 8; ++v) {
    float inv = 1.0f / lrow[v];
    size_t row = (size_t)(b * S_ + qt * 16 + v + 8 * half) * D_ + h * HD_;
    out[row + 0  + lw] = (__bf16)(acc0[v] * inv);
    out[row + 16 + lw] = (__bf16)(acc1[v] * inv);
    out[row + 32 + lw] = (__bf16)(acc2[v] * inv);
    out[row + 48 + lw] = (__bf16)(acc3[v] * inv);
  }
}

// ---------------------------------------------------------------------------
// Host-side orchestration
// ---------------------------------------------------------------------------
extern "C" void kernel_launch(void* const* d_in, const int* in_sizes, int n_in,
                              void* d_out, int out_size, void* d_ws,
                              size_t ws_size, hipStream_t stream) {
  (void)in_sizes; (void)n_in; (void)out_size; (void)ws_size;

  const int*   tok    = (const int*)d_in[0];
  const int*   nup    = (const int*)d_in[1];
  const int*   npad   = (const int*)d_in[2];
  const int*   nch    = (const int*)d_in[3];
  const float* emb    = (const float*)d_in[4];
  const float* proj_w = (const float*)d_in[5];
  const float* proj_b = (const float*)d_in[6];
  // blocks[i] starts at d_in[7 + 12*i]:
  //   in_w,in_b,out_w,out_b,ln1_g,ln1_b,ff1_w,ff1_b,ff2_w,ff2_b,ln2_g,ln2_b

  size_t off = 0;
  auto carve = [&](size_t bytes) -> void* {
    void* p = (char*)d_ws + off;
    off += (bytes + 255) & ~(size_t)255;
    return p;
  };
  float*  feat  = (float*)carve((size_t)M_ * D_ * 4);
  float*  tmp   = (float*)carve((size_t)M_ * D_ * 4);
  __bf16* xb    = (__bf16*)carve((size_t)M_ * D_ * 2);
  __bf16* qkvb  = (__bf16*)carve((size_t)M_ * QKV_ * 2);
  __bf16* vtb   = (__bf16*)carve((size_t)B_ * H_ * HD_ * S_ * 2);
  __bf16* attnb = (__bf16*)carve((size_t)M_ * D_ * 2);
  __bf16* ff1b  = (__bf16*)carve((size_t)M_ * FF_ * 2);

  const int n_inw = QKV_ * D_, n_outw = D_ * D_;
  const int n_ff1 = FF_ * D_,  n_ff2  = D_ * FF_;
  __bf16* wb_in[NB_];
  __bf16* wb_out[NB_];
  __bf16* wb_f1[NB_];
  __bf16* wb_f2[NB_];
  for (int i = 0; i < NB_; ++i) {
    wb_in[i]  = (__bf16*)carve((size_t)n_inw * 2);
    wb_out[i] = (__bf16*)carve((size_t)n_outw * 2);
    wb_f1[i]  = (__bf16*)carve((size_t)n_ff1 * 2);
    wb_f2[i]  = (__bf16*)carve((size_t)n_ff2 * 2);
  }

  // --- weight conversion to bf16 ---
  for (int i = 0; i < NB_; ++i) {
    const float* in_w  = (const float*)d_in[7 + 12 * i + 0];
    const float* out_w = (const float*)d_in[7 + 12 * i + 2];
    const float* ff1_w = (const float*)d_in[7 + 12 * i + 6];
    const float* ff2_w = (const float*)d_in[7 + 12 * i + 8];
    convert_bf16_kernel<<<(n_inw + 255) / 256, 256, 0, stream>>>(in_w, wb_in[i], n_inw);
    convert_bf16_kernel<<<(n_outw + 255) / 256, 256, 0, stream>>>(out_w, wb_out[i], n_outw);
    convert_bf16_kernel<<<(n_ff1 + 255) / 256, 256, 0, stream>>>(ff1_w, wb_f1[i], n_ff1);
    convert_bf16_kernel<<<(n_ff2 + 255) / 256, 256, 0, stream>>>(ff2_w, wb_f2[i], n_ff2);
  }

  // --- embedding ---
  embed_kernel<<<(int)(((size_t)M_ * D_ + 255) / 256), 256, 0, stream>>>(
      tok, nup, emb, feat, xb);

  // --- transformer blocks ---
  for (int i = 0; i < NB_; ++i) {
    const float* in_b  = (const float*)d_in[7 + 12 * i + 1];
    const float* out_b = (const float*)d_in[7 + 12 * i + 3];
    const float* ln1_g = (const float*)d_in[7 + 12 * i + 4];
    const float* ln1_b = (const float*)d_in[7 + 12 * i + 5];
    const float* ff1_b = (const float*)d_in[7 + 12 * i + 7];
    const float* ff2_b = (const float*)d_in[7 + 12 * i + 9];
    const float* ln2_g = (const float*)d_in[7 + 12 * i + 10];
    const float* ln2_b = (const float*)d_in[7 + 12 * i + 11];

    // QKV projection: (M,768) = X(M,256) @ in_w(768,256)^T
    gemm_wmma_kernel<<<(M_ / 16) * (QKV_ / 64) / 8, 256, 0, stream>>>(
        xb, wb_in[i], in_b, nullptr, qkvb, M_, QKV_, D_, 0);

    transpose_v_kernel<<<(int)(((size_t)B_ * H_ * HD_ * S_ + 255) / 256), 256,
                         0, stream>>>(qkvb, vtb);

    attention_kernel<<<(B_ * H_ * (S_ / 16)) / 8, 256, 0, stream>>>(
        qkvb, vtb, npad, nch, attnb);

    // output projection
    gemm_wmma_kernel<<<(M_ / 16) * (D_ / 64) / 8, 256, 0, stream>>>(
        attnb, wb_out[i], out_b, tmp, nullptr, M_, D_, D_, 0);
    add_ln_kernel<<<M_, 256, 0, stream>>>(feat, tmp, ln1_g, ln1_b, feat, xb);

    // FFN
    gemm_wmma_kernel<<<(M_ / 16) * (FF_ / 64) / 8, 256, 0, stream>>>(
        xb, wb_f1[i], ff1_b, nullptr, ff1b, M_, FF_, D_, 1);
    gemm_wmma_kernel<<<(M_ / 16) * (D_ / 64) / 8, 256, 0, stream>>>(
        ff1b, wb_f2[i], ff2_b, tmp, nullptr, M_, D_, FF_, 0);
    add_ln_kernel<<<M_, 256, 0, stream>>>(feat, tmp, ln2_g, ln2_b, feat, xb);
  }

  // --- final 2-way projection ---
  proj_kernel<<<(M_ + 255) / 256, 256, 0, stream>>>(feat, proj_w, proj_b,
                                                    (float*)d_out);
}
